// InfoNCECauchySelfSupervised_8572754722863
// MI455X (gfx1250) — matrix operand order repaired and checked by reading, not compiled, not run on hardware
//
#include <hip/hip_runtime.h>
#include <hip/hip_bf16.h>
#include <cstdint>

// ---------------------------------------------------------------------------
// InfoNCE Cauchy self-supervised loss for MI455X (gfx1250, wave32, WMMA).
// features: (8192,128) fp32. b = 4096.
//   K(x,y) = 1 / (1 + max(|x|^2+|y|^2-2 x.y, 0))
//   loss = (1/b) * sum_i [ 0.5*log(S11_i - 1 + R12_i)
//                        + 0.5*log(C12_i + S22_i - 1) - log(D12_i) ]
// Gram dots via v_wmma_f32_16x16x32_bf16 with bf16 hi/lo split (3 passes).
// z11/z22 are symmetric: only j-tile >= i-tile computed; mirrored tile row
// sums recovered from column sums. Tile staging uses CDNA5 async-to-LDS.
// ---------------------------------------------------------------------------

typedef __attribute__((ext_vector_type(16))) __bf16 v16bf;
typedef __attribute__((ext_vector_type(8)))  __bf16 v8bf;
typedef __attribute__((ext_vector_type(8)))  float  v8f;

#define BHALF 4096
#define KDIM  128
#define KP    136   // padded LDS row stride (272 B: 16B-aligned, bank-shifted)

__device__ __forceinline__ uint16_t f32_to_bf16_rne(float f) {
    union { float f; uint32_t u; } c; c.f = f;
    uint32_t u = c.u;
    uint32_t r = u + 0x7FFFu + ((u >> 16) & 1u);   // round-to-nearest-even
    return (uint16_t)(r >> 16);
}
__device__ __forceinline__ float bf16_bits_to_f32(uint16_t h) {
    union { uint32_t u; float f; } c; c.u = ((uint32_t)h) << 16;
    return c.f;
}

// async copy of one 16B chunk: LDS[ldsaddr] = MEM[gaddr]  (ASYNCcnt-tracked)
__device__ __forceinline__ void async_b128(uint32_t ldsaddr, uint64_t gaddr) {
    asm volatile("global_load_async_to_lds_b128 %0, %1, off"
                 :: "v"(ldsaddr), "v"(gaddr) : "memory");
}

// --- 1) split fp32 rows into bf16 hi/lo + fp32 squared norms (1 wave/row) ---
__global__ __launch_bounds__(256)
void prep_kernel(const float* __restrict__ feats,
                 uint16_t* __restrict__ Ahi, uint16_t* __restrict__ Alo,
                 float* __restrict__ sq) {
    const int wid  = threadIdx.x >> 5;
    const int lane = threadIdx.x & 31;
    const int row  = blockIdx.x * 8 + wid;
    const float* x = feats + (size_t)row * KDIM;
    float s = 0.f;
#pragma unroll
    for (int j = 0; j < 4; ++j) {
        const int k = lane + 32 * j;
        const float v = x[k];
        const uint16_t h = f32_to_bf16_rne(v);
        const float hf = bf16_bits_to_f32(h);
        Ahi[(size_t)row * KDIM + k] = h;
        Alo[(size_t)row * KDIM + k] = f32_to_bf16_rne(v - hf);
        s = fmaf(v, v, s);
    }
#pragma unroll
    for (int off = 16; off > 0; off >>= 1) s += __shfl_xor(s, off, 32);
    if (lane == 0) sq[row] = s;
}

// --- 2) exact fp32 diag of z12 (feeds log() directly) -----------------------
__global__ __launch_bounds__(256)
void diag_kernel(const float* __restrict__ feats, const float* __restrict__ sq,
                 float* __restrict__ D12) {
    const int wid  = threadIdx.x >> 5;
    const int lane = threadIdx.x & 31;
    const int i    = blockIdx.x * 8 + wid;
    const float* x = feats + (size_t)i * KDIM;
    const float* y = feats + (size_t)(BHALF + i) * KDIM;
    float dot = 0.f;
#pragma unroll
    for (int j = 0; j < 4; ++j) {
        const int k = lane + 32 * j;
        dot = fmaf(x[k], y[k], dot);
    }
#pragma unroll
    for (int off = 16; off > 0; off >>= 1) dot += __shfl_xor(dot, off, 32);
    if (lane == 0) {
        float d2 = fmaxf(sq[i] + sq[BHALF + i] - 2.f * dot, 0.f);
        D12[i] = 1.f / (1.f + d2);
    }
}

// --- 3) WMMA Gram tiles: 64x64 per block, z = {z11, z22, z12} ---------------
__global__ __launch_bounds__(128)
void gram_kernel(const uint16_t* __restrict__ Ahi, const uint16_t* __restrict__ Alo,
                 const float* __restrict__ sq,
                 float* __restrict__ S11, float* __restrict__ S22,
                 float* __restrict__ R12, float* __restrict__ C12) {
    __shared__ __align__(128) uint16_t smem[4 * 64 * KP];   // ~68 KB
    uint16_t* sAhi = smem;
    uint16_t* sAlo = smem + 1 * 64 * KP;
    uint16_t* sBhi = smem + 2 * 64 * KP;
    uint16_t* sBlo = smem + 3 * 64 * KP;
    float* sRow = (float*)smem;          // aliased after compute phase
    float* sCol = (float*)smem + 64;

    const int t  = blockIdx.z;           // 0:z11  1:z22  2:z12
    const int i0 = blockIdx.x * 64;
    const int j0 = blockIdx.y * 64;
    if (t < 2 && j0 < i0) return;        // symmetric: mirrored tile is implied

    const int arow0 = (t == 1 ? BHALF : 0) + i0;
    const int brow0 = (t == 0 ? 0 : BHALF) + j0;
    const int tid = threadIdx.x;

    // row sums always accumulate at i0; column sums (z12, or the mirrored
    // rows of a symmetric off-diagonal tile) accumulate at j0.
    float* rowArr = (t == 0) ? S11 : (t == 1) ? S22 : R12;
    float* colArr = (t == 2) ? C12 : rowArr;
    const bool doCols = (t == 2) || (i0 != j0);

    // Stage 64x128 hi/lo tiles with async global->LDS copies (no VGPR bounce).
    {
        const uint32_t lA = (uint32_t)(uintptr_t)sAhi;
        const uint32_t lB = (uint32_t)(uintptr_t)sAlo;
        const uint32_t lC = (uint32_t)(uintptr_t)sBhi;
        const uint32_t lD = (uint32_t)(uintptr_t)sBlo;
        const uint64_t gA = (uint64_t)(uintptr_t)(Ahi + (size_t)arow0 * KDIM);
        const uint64_t gB = (uint64_t)(uintptr_t)(Alo + (size_t)arow0 * KDIM);
        const uint64_t gC = (uint64_t)(uintptr_t)(Ahi + (size_t)brow0 * KDIM);
        const uint64_t gD = (uint64_t)(uintptr_t)(Alo + (size_t)brow0 * KDIM);
        for (int u = tid; u < 1024; u += 128) {            // 64 rows * 16 chunks
            const int row = u >> 4, c = u & 15;
            const uint32_t doff = (uint32_t)(row * (KP * 2) + c * 16);
            const uint64_t goff = (uint64_t)u * 16;
            async_b128(lA + doff, gA + goff);
            async_b128(lB + doff, gB + goff);
            async_b128(lC + doff, gC + goff);
            async_b128(lD + doff, gD + goff);
        }
        asm volatile("s_wait_asynccnt 0" ::: "memory");
    }
    __syncthreads();

    const int w    = tid >> 5;       // wave -> 16-row strip
    const int lane = tid & 31;
    const int ln   = lane & 15;
    const int hi   = lane >> 4;

    v8f acc[4] = {};                 // 4 subtiles of 16x16 f32 per wave

    const uint16_t* pa[3] = { sAhi, sAhi, sAlo };   // hi*hi + hi*lo + lo*hi
    const uint16_t* pb[3] = { sBhi, sBlo, sBhi };
#pragma unroll
    for (int p = 0; p < 3; ++p) {
        const uint16_t* arow = pa[p] + (size_t)(16 * w + ln) * KP;
#pragma unroll
        for (int ks = 0; ks < 4; ++ks) {
            const int k0 = 32 * ks;
            // A 16x32 bf16 layout: K = 8*(L/16)+0..7 then 16+8*(L/16)+0..7
            v8bf a0 = *(const v8bf*)(arow + k0 + 8 * hi);
            v8bf a1 = *(const v8bf*)(arow + k0 + 16 + 8 * hi);
            v16bf af = __builtin_shufflevector(a0, a1,
                0,1,2,3,4,5,6,7,8,9,10,11,12,13,14,15);
#pragma unroll
            for (int sj = 0; sj < 4; ++sj) {
                // B 32x16 bf16 layout: K = 8*(L/16)+0..7 then 16+8*(L/16)+0..7
                const uint16_t* brow =
                    pb[p] + (size_t)(16 * sj + ln) * KP + k0 + 8 * hi;
                v8bf b0 = *(const v8bf*)(brow);
                v8bf b1 = *(const v8bf*)(brow + 16);
                v16bf bf = __builtin_shufflevector(b0, b1,
                    0,1,2,3,4,5,6,7,8,9,10,11,12,13,14,15);
                acc[sj] = __builtin_amdgcn_wmma_f32_16x16x32_bf16(
                    false, af, false, bf, (short)0, acc[sj], false, false);
            }
        }
    }

    __syncthreads();                  // staging LDS dead; alias sum arrays
    if (tid < 64) sCol[tid] = 0.f;
    __syncthreads();

    float nsq[4];
#pragma unroll
    for (int sj = 0; sj < 4; ++sj) nsq[sj] = sq[brow0 + 16 * sj + ln];

    float colpart[4] = {0.f, 0.f, 0.f, 0.f};
#pragma unroll
    for (int r = 0; r < 8; ++r) {
        const int mloc = 16 * w + r + 8 * hi;     // C/D layout: M = r + 8*(L/16)
        const float msq = sq[arow0 + mloc];
        float rowpart = 0.f;
#pragma unroll
        for (int sj = 0; sj < 4; ++sj) {
            float d2 = fmaxf(msq + nsq[sj] - 2.f * acc[sj][r], 0.f);
            const float kv = 1.f / (1.f + d2);
            rowpart += kv;
            colpart[sj] += kv;
        }
#pragma unroll
        for (int off = 1; off < 16; off <<= 1)
            rowpart += __shfl_xor(rowpart, off, 16);
        if (ln == 0) sRow[mloc] = rowpart;        // unique writer per row
    }
    if (doCols) {
#pragma unroll
        for (int sj = 0; sj < 4; ++sj) {
            float cp = colpart[sj] + __shfl_xor(colpart[sj], 16, 32);
            if (lane < 16) atomicAdd(&sCol[16 * sj + ln], cp);
        }
    }
    __syncthreads();

    if (tid < 64) {
        atomicAdd(rowArr + i0 + tid, sRow[tid]);
        if (doCols) atomicAdd(colArr + j0 + tid, sCol[tid]);
    }
}

// --- 4) fold 5 length-4096 vectors into the scalar loss ---------------------
__global__ __launch_bounds__(256)
void reduce_kernel(const float* __restrict__ S11, const float* __restrict__ S22,
                   const float* __restrict__ R12, const float* __restrict__ C12,
                   const float* __restrict__ D12, float* __restrict__ out) {
    const int tid = threadIdx.x;
    float s = 0.f;
    for (int i = tid; i < BHALF; i += 256) {
        const float a = (S11[i] - 1.0f) + R12[i];
        const float c = C12[i] + (S22[i] - 1.0f);
        s += 0.5f * logf(a) + 0.5f * logf(c) - logf(D12[i]);
    }
#pragma unroll
    for (int off = 16; off > 0; off >>= 1) s += __shfl_xor(s, off, 32);
    __shared__ float part[8];
    if ((tid & 31) == 0) part[tid >> 5] = s;
    __syncthreads();
    if (tid == 0) {
        float tot = 0.f;
        for (int k = 0; k < 8; ++k) tot += part[k];
        out[0] = tot / (float)BHALF;
    }
}

extern "C" void kernel_launch(void* const* d_in, const int* in_sizes, int n_in,
                              void* d_out, int out_size, void* d_ws, size_t ws_size,
                              hipStream_t stream) {
    const float* feats = (const float*)d_in[0];   // (8192,128) fp32; labels unused
    char* ws = (char*)d_ws;

    uint16_t* Ahi = (uint16_t*)ws;                          // 2 MB
    uint16_t* Alo = (uint16_t*)(ws + (size_t)(2u << 20));   // 2 MB
    float*    sq  = (float*)(ws + (size_t)(4u << 20));      // 32 KB
    float*    S11 = (float*)(ws + (size_t)(4u << 20) + (32u << 10));
    float*    S22 = S11 + 1 * BHALF;
    float*    R12 = S11 + 2 * BHALF;
    float*    C12 = S11 + 3 * BHALF;
    float*    D12 = S11 + 4 * BHALF;

    hipMemsetAsync(S11, 0, (size_t)4 * BHALF * sizeof(float), stream);

    prep_kernel<<<1024, 256, 0, stream>>>(feats, Ahi, Alo, sq);
    diag_kernel<<<512, 256, 0, stream>>>(feats, sq, D12);
    gram_kernel<<<dim3(64, 64, 3), 128, 0, stream>>>(Ahi, Alo, sq, S11, S22, R12, C12);
    reduce_kernel<<<1, 256, 0, stream>>>(S11, S22, R12, C12, D12, (float*)d_out);
}
